// OccViT_41609643164387
// MI455X (gfx1250) — matrix operand order repaired and checked by reading, not compiled, not run on hardware
//
#include <hip/hip_runtime.h>

// CDNA5 (gfx1250) wave32 WMMA fragment types
typedef __attribute__((ext_vector_type(16))) _Float16 v16h;
typedef __attribute__((ext_vector_type(8)))  _Float16 v8h;
typedef __attribute__((ext_vector_type(8)))  float    v8f;

#define SEQ 9          // valid token rows (cls + 8 views); padded to 16 for WMMA
#define XS  1544       // xbuf row stride (halves); 3088B = 16B-aligned rows
#define HS  264        // hbuf row stride (halves); 528B = 16B-aligned rows
#define YS  260        // y row stride (floats)

// ---- workspace layout (halves) for fragment-major f16 weights ----
#define OFF_VIEW 0u                 // 320x256  -> 160 tiles
#define OFF_CLS  81920u             // 320x256
#define OFF_QKV  163840u            // 4 x (256x1536) -> 768 tiles/layer
#define OFF_OUT  1736704u           // 4 x (512x256)  -> 256 tiles/layer
#define OFF_FF1  2260992u           // 4 x (256x1024) -> 512 tiles/layer
#define OFF_FF2  3309568u           // 4 x (1024x256) -> 512 tiles/layer
#define OFF_HW1  4358144u           // 256x256 -> 128 tiles
#define OFF_HW2  4423680u           // 256x256
#define WS_HALVES 4489216u          // ~8.98 MB

__device__ __forceinline__ v8f wmma32(v16h a, v16h b, v8f c) {
    return __builtin_amdgcn_wmma_f32_16x16x32_f16(false, a, false, b,
                                                  (short)0, c, false, false);
}

// 32B LDS vector load via two 16B halves (addresses are 16B aligned)
__device__ __forceinline__ v16h load16h(const _Float16* p) {
    v8h lo = *(const v8h*)p;
    v8h hi = *(const v8h*)(p + 8);
    v16h r;
    #pragma unroll
    for (int e = 0; e < 8; ++e) { r[e] = lo[e]; r[8 + e] = hi[e]; }
    return r;
}

// ---- A fragment: 16x32 f16 from LDS (row-major, stride lda halves) ----
// Lanes 0-15 -> M=lane ; lanes 16-31 -> M=lane-16, K+16.
// elems 0..15 -> K = hh*16+0..15 (contiguous).
// Rows >= SEQ are CLAMPED (branchless): garbage rows only pollute D rows >= SEQ,
// which every D-store masks. No EXEC divergence in the WMMA loop.
__device__ __forceinline__ v16h load_a_frag(const _Float16* A, int lda, int k0) {
    int lane = threadIdx.x & 31;
    int row  = lane & 15;
    row = (row < SEQ) ? row : (SEQ - 1);
    int hh = (lane >> 4) << 4;
    const _Float16* p = A + row * lda + k0 + hh;
    v16h a;
    #pragma unroll
    for (int e = 0; e < 8; ++e) { a[e] = p[e]; a[8 + e] = p[8 + e]; }
    return a;
}

// ---- B fragment from prepped fragment-major f16 weights ----
// tile = (n0/16)*kTiles + k0/32 ; lane's 16 halves are contiguous (32B load).
__device__ __forceinline__ v16h load_b_prep(const _Float16* Wp, int kTiles, int k0, int n0) {
    int lane = threadIdx.x & 31;
    const _Float16* p = Wp + ((size_t)(n0 >> 4) * kTiles + (k0 >> 5)) * 512 + lane * 16;
    return *(const v16h*)p;   // 32B aligned: tiles are 1KB aligned
}

__device__ __forceinline__ v8f gemm_tile_p(const _Float16* A, int lda,
                                           const _Float16* Wp, int K, int n0) {
    int kT = K >> 5;
    v8f c = {};
    for (int k0 = 0; k0 < K; k0 += 32) {
        v16h a = load_a_frag(A, lda, k0);
        v16h b = load_b_prep(Wp, kT, k0, n0);
        c = wmma32(a, b, c);
    }
    return c;
}

// D layout: col = n0 + (lane&15), rows = i + 8*(lane>>4)
__device__ __forceinline__ void store_add_y(v8f c, float* yb, int n0, const float* bias) {
    int lane = threadIdx.x & 31;
    int col  = n0 + (lane & 15);
    int r0   = (lane >> 4) << 3;
    float bv = bias[col];
    #pragma unroll
    for (int i = 0; i < 8; ++i) {
        int row = r0 + i;
        if (row < SEQ) yb[row * YS + col] += c[i] + bv;
    }
}

__device__ __forceinline__ void store_x16(v8f c, _Float16* ob, int ldo, int n0,
                                          const float* bias, bool gelu) {
    int lane = threadIdx.x & 31;
    int col  = n0 + (lane & 15);
    int r0   = (lane >> 4) << 3;
    float bv = bias ? bias[col] : 0.0f;
    #pragma unroll
    for (int i = 0; i < 8; ++i) {
        int row = r0 + i;
        float v = c[i] + bv;
        if (gelu) v = 0.5f * v * (1.0f + erff(v * 0.70710678118654752f));
        if (row < SEQ) ob[row * ldo + col] = (_Float16)v;
    }
}

__device__ __forceinline__ void store_embed(v8f c, float* yb, int n0) {   // rows 1..8
    int lane = threadIdx.x & 31;
    int col  = n0 + (lane & 15);
    int r0   = (lane >> 4) << 3;
    #pragma unroll
    for (int i = 0; i < 8; ++i) {
        int row = r0 + i;
        if (row >= 1 && row < SEQ) yb[row * YS + col] = c[i];
    }
}

__device__ __forceinline__ void store_row0_y(v8f c, float* yb, int n0) {  // row 0 only
    int lane = threadIdx.x & 31;
    if (lane < 16) yb[n0 + lane] = c[0];
}

__device__ __forceinline__ void store_row0_h16_relu(v8f c, _Float16* ob, int n0,
                                                    const float* bias) {
    int lane = threadIdx.x & 31;
    if (lane < 16) {
        int col = n0 + lane;
        ob[col] = (_Float16)fmaxf(c[0] + bias[col], 0.0f);
    }
}

__device__ __forceinline__ void store_row0_f32_relu(v8f c, float* zb, int n0,
                                                    const float* bias) {
    int lane = threadIdx.x & 31;
    if (lane < 16) {
        int col = n0 + lane;
        zb[col] = fmaxf(c[0] + bias[col], 0.0f);
    }
}

// LayerNorm over T=256 for rows 0..8: 16 threads/row, shfl_xor reduce (width 16)
__device__ __forceinline__ void ln_rows(const float* yb, _Float16* hb,
                                        const float* g, const float* b) {
    int tid = threadIdx.x;
    if (tid < 16 * SEQ) {
        int row = tid >> 4;
        int c0  = (tid & 15) * 16;
        float s = 0.0f, sq = 0.0f;
        #pragma unroll
        for (int j = 0; j < 16; ++j) {
            float v = yb[row * YS + c0 + j];
            s += v; sq += v * v;
        }
        #pragma unroll
        for (int m = 1; m < 16; m <<= 1) {
            s  += __shfl_xor(s,  m, 16);
            sq += __shfl_xor(sq, m, 16);
        }
        float mean = s * (1.0f / 256.0f);
        float var  = sq * (1.0f / 256.0f) - mean * mean;
        float rs   = rsqrtf(var + 1e-5f);
        #pragma unroll
        for (int j = 0; j < 16; ++j) {
            int c = c0 + j;
            hb[row * HS + c] = (_Float16)((yb[row * YS + c] - mean) * rs * g[c] + b[c]);
        }
    }
}

// ---------- weight prep: f32 [K x N] -> fragment-major f16 tiles ----------
__global__ __launch_bounds__(256)
void prep_weights(const float* __restrict__ src, _Float16* __restrict__ dst,
                  int K, int N, size_t srcStride, size_t dstStride) {
    const float* s = src + (size_t)blockIdx.y * srcStride;
    _Float16*    o = dst + (size_t)blockIdx.y * dstStride;
    int kT = K >> 5;
    int ntiles = kT * (N >> 4);
    int g = blockIdx.x * 256 + threadIdx.x;
    int tile = g >> 5, lane = g & 31;
    if (tile >= ntiles) return;
    int nt  = tile / kT, kt = tile % kT;
    int col = (nt << 4) + (lane & 15);
    int kr  = (kt << 5) + ((lane >> 4) << 4);
    v16h v;
    #pragma unroll
    for (int e = 0; e < 16; ++e) v[e] = (_Float16)s[(size_t)(kr + e) * N + col];
    *(v16h*)(o + (size_t)tile * 512 + lane * 16) = v;
}

// ------------------------------ fused model ------------------------------
__global__ __launch_bounds__(256)
void occvit_fused(const float* __restrict__ img_feat,     // [N,8,256]
                  const float* __restrict__ pts_view,     // [N,8,64]
                  const float* __restrict__ pts_world,    // [N,64]
                  const float* __restrict__ view_b,
                  const float* __restrict__ cls_token,
                  const float* __restrict__ cls_b,
                  const float* __restrict__ pos_emb,      // [9,256]
                  const float* __restrict__ ln1_g, const float* __restrict__ ln1_b,
                  const float* __restrict__ reattn_w,     // [4,8,8]
                  const float* __restrict__ reattn_g, const float* __restrict__ reattn_b,
                  const float* __restrict__ out_b,
                  const float* __restrict__ ln2_g, const float* __restrict__ ln2_b,
                  const float* __restrict__ ff_b1, const float* __restrict__ ff_b2,
                  const float* __restrict__ head_ln_g, const float* __restrict__ head_ln_b,
                  const float* __restrict__ head_b1, const float* __restrict__ head_b2,
                  const float* __restrict__ head_w3, const float* __restrict__ head_b3,
                  const _Float16* __restrict__ wsp,       // prepped f16 weights
                  float* __restrict__ outp) {
    __shared__ __align__(32) float     y_s[SEQ * YS];
    __shared__ __align__(32) _Float16  xbuf[SEQ * XS];
    __shared__ __align__(32) _Float16  hbuf[SEQ * HS];
    __shared__ __align__(32) _Float16  abuf[8 * 16 * 32];  // attn staged, zero-padded
    __shared__ float attnA[8 * 9 * 9];
    __shared__ float attnB[8 * 9 * 9];
    __shared__ float red[32];

    const int n    = blockIdx.x;
    const int tid  = threadIdx.x;
    const int wave = tid >> 5;
    const int lane = tid & 31;

    const _Float16* wView = wsp + OFF_VIEW;
    const _Float16* wCls  = wsp + OFF_CLS;

    // ---------------- embedding: build X [9 x 320] (f16) ----------------
    for (int idx = tid; idx < SEQ * 320; idx += 256) {
        int s = idx / 320, c = idx % 320;
        float v;
        if (s == 0) {
            v = (c < 256) ? cls_token[c] : pts_world[(size_t)n * 64 + (c - 256)];
        } else {
            int vv = s - 1;
            v = (c < 256) ? img_feat[((size_t)n * 8 + vv) * 256 + c]
                          : pts_view[((size_t)n * 8 + vv) * 64 + (c - 256)];
        }
        xbuf[s * XS + c] = (_Float16)v;
    }
    __syncthreads();

    // rows 1..8 via view_w, row 0 via cls_w (all WMMA)
    for (int t = wave; t < 16; t += 8) {
        v8f c = gemm_tile_p(xbuf, XS, wView, 320, t * 16);
        store_embed(c, y_s, t * 16);
        v8f c0 = gemm_tile_p(xbuf, XS, wCls, 320, t * 16);
        store_row0_y(c0, y_s, t * 16);
    }
    __syncthreads();
    for (int idx = tid; idx < SEQ * 256; idx += 256) {
        int s = idx >> 8, c = idx & 255;
        y_s[s * YS + c] += (s == 0 ? cls_b[c] : view_b[c]) + pos_emb[s * 256 + c];
    }
    __syncthreads();

    // ---------------- transformer layers ----------------
    for (int d = 0; d < 4; ++d) {
        const _Float16* wQKV = wsp + OFF_QKV + (size_t)d * 393216;
        const _Float16* wOut = wsp + OFF_OUT + (size_t)d * 131072;
        const _Float16* wF1  = wsp + OFF_FF1 + (size_t)d * 262144;
        const _Float16* wF2  = wsp + OFF_FF2 + (size_t)d * 262144;
        const float* rw = reattn_w + d * 64;
        const float* rg = reattn_g + d * 8;
        const float* rb = reattn_b + d * 8;

        if (d < 3 && tid == 0) {   // warm next layer's weights toward L2
            __builtin_prefetch(wsp + OFF_QKV + (size_t)(d + 1) * 393216, 0, 1);
            __builtin_prefetch(wsp + OFF_FF1 + (size_t)(d + 1) * 262144, 0, 1);
        }

        // LN1 -> hbuf
        ln_rows(y_s, hbuf, ln1_g + d * 256, ln1_b + d * 256);
        __syncthreads();

        // QKV: [16x256] @ [256x1536] -> xbuf (96 tiles)
        for (int t = wave; t < 96; t += 8) {
            v8f c = gemm_tile_p(hbuf, HS, wQKV, 256, t * 16);
            store_x16(c, xbuf, XS, t * 16, nullptr, false);
        }
        __syncthreads();

        // dots = q @ k^T per head, via WMMA (one head per wave; K=64)
        {
            int h = wave;
            v8f c = {};
            for (int k0 = 0; k0 < 64; k0 += 32) {
                v16h a = load_a_frag(xbuf + h * 64, XS, k0);
                int j  = lane & 15;
                int jj = (j < SEQ) ? j : (SEQ - 1);
                // B[kk][j] = k_mat[j][kk] : contiguous halves along kk in row j
                v16h b = load16h(xbuf + jj * XS + 512 + h * 64 + k0 + ((lane >> 4) << 4));
                c = wmma32(a, b, c);
            }
            int j  = lane & 15;
            int r0 = (lane >> 4) << 3;
            if (j < SEQ) {
                #pragma unroll
                for (int i = 0; i < 8; ++i) {
                    int row = r0 + i;
                    if (row < SEQ) attnA[(h * 9 + row) * 9 + j] = c[i] * 0.125f;
                }
            }
        }
        __syncthreads();

        // softmax over j (72 rows of 9)
        if (tid < 72) {
            int h = tid / 9, i = tid % 9;
            float* a = attnA + (h * 9 + i) * 9;
            float m = -1e30f;
            for (int j = 0; j < 9; ++j) m = fmaxf(m, a[j]);
            float s = 0.0f;
            for (int j = 0; j < 9; ++j) { float e = __expf(a[j] - m); a[j] = e; s += e; }
            float inv = 1.0f / s;
            for (int j = 0; j < 9; ++j) a[j] *= inv;
        }
        __syncthreads();

        // re-attention: attnB[g,i,j] = sum_h attnA[h,i,j] * rw[h,g]
        for (int idx = tid; idx < 648; idx += 256) {
            int g = idx / 81, r = idx % 81, i = r / 9, j = r % 9;
            float acc = 0.0f;
            #pragma unroll
            for (int h = 0; h < 8; ++h) acc += attnA[(h * 9 + i) * 9 + j] * rw[h * 8 + g];
            attnB[(g * 9 + i) * 9 + j] = acc;
        }
        __syncthreads();

        // LayerNorm over head axis, result back to attnA
        if (tid < 81) {
            int i = tid / 9, j = tid % 9;
            float s = 0.0f, sq = 0.0f;
            #pragma unroll
            for (int g = 0; g < 8; ++g) {
                float v = attnB[(g * 9 + i) * 9 + j];
                s += v; sq += v * v;
            }
            float mean = s * 0.125f;
            float var  = sq * 0.125f - mean * mean;
            float rs   = rsqrtf(var + 1e-5f);
            #pragma unroll
            for (int g = 0; g < 8; ++g) {
                float v = attnB[(g * 9 + i) * 9 + j];
                attnA[(g * 9 + i) * 9 + j] = (v - mean) * rs * rg[g] + rb[g];
            }
        }
        __syncthreads();

        // stage attn as zero-padded f16 [8][16][32] (zeros kill K>=9 garbage)
        for (int idx = tid; idx < 8 * 16 * 32; idx += 256) {
            int h = idx >> 9, i = (idx >> 5) & 15, j = idx & 31;
            float v = (i < SEQ && j < SEQ) ? attnA[(h * 9 + i) * 9 + j] : 0.0f;
            abuf[idx] = (_Float16)v;
        }
        __syncthreads();

        // attn @ v -> xbuf[:, 0:512] via WMMA (32 tiles: 8 heads x 4 d-tiles, K=32)
        for (int t = wave; t < 32; t += 8) {
            int h = t >> 2, dt = t & 3;
            v16h a = load_a_frag(abuf + h * 512, 32, 0);
            int colD = (h << 6) + (dt << 4) + (lane & 15);
            v16h b;
            #pragma unroll
            for (int e = 0; e < 16; ++e) {
                int j  = ((lane >> 4) << 4) + e;
                int jj = (j < SEQ) ? j : (SEQ - 1);
                b[e] = xbuf[jj * XS + 1024 + colD];
            }
            v8f c = {};
            c = wmma32(a, b, c);
            store_x16(c, xbuf, XS, (h << 6) + (dt << 4), nullptr, false);
        }
        __syncthreads();

        // out proj: y += [16x512] @ [512x256] + out_b
        for (int t = wave; t < 16; t += 8) {
            v8f c = gemm_tile_p(xbuf, XS, wOut, 512, t * 16);
            store_add_y(c, y_s, t * 16, out_b + d * 256);
        }
        __syncthreads();

        // LN2 -> hbuf
        ln_rows(y_s, hbuf, ln2_g + d * 256, ln2_b + d * 256);
        __syncthreads();

        // FFN1 + exact GELU
        for (int t = wave; t < 64; t += 8) {
            v8f c = gemm_tile_p(hbuf, HS, wF1, 256, t * 16);
            store_x16(c, xbuf, XS, t * 16, ff_b1 + d * 1024, true);
        }
        __syncthreads();

        // FFN2: y += [16x1024] @ [1024x256] + ff_b2
        for (int t = wave; t < 16; t += 8) {
            v8f c = gemm_tile_p(xbuf, XS, wF2, 1024, t * 16);
            store_add_y(c, y_s, t * 16, ff_b2 + d * 256);
        }
        __syncthreads();
    }

    // ---------------- head: LN(cls) -> WMMA relu MLP -> scalar ----------------
    float* zA = attnA;   // f32 scratch (648 >= 256)
    {
        float val = y_s[tid];
        float s = val, sq = val * val;
        #pragma unroll
        for (int m = 1; m < 32; m <<= 1) {
            s  += __shfl_xor(s,  m, 32);
            sq += __shfl_xor(sq, m, 32);
        }
        if ((tid & 31) == 0) { red[wave] = s; red[8 + wave] = sq; }
        __syncthreads();
        if (tid == 0) {
            float ts = 0.0f, tq = 0.0f;
            for (int w = 0; w < 8; ++w) { ts += red[w]; tq += red[8 + w]; }
            float mean = ts * (1.0f / 256.0f);
            float var  = tq * (1.0f / 256.0f) - mean * mean;
            red[16] = mean;
            red[17] = rsqrtf(var + 1e-5f);
        }
        __syncthreads();
        float z = (val - red[16]) * red[17] * head_ln_g[tid] + head_ln_b[tid];
        hbuf[tid] = (_Float16)z;       // row 0 of hbuf (rows>=1 stale -> masked)
    }
    __syncthreads();
    // z @ head_w1, relu -> xbuf row 0
    for (int t = wave; t < 16; t += 8) {
        v8f c = gemm_tile_p(hbuf, HS, wsp + OFF_HW1, 256, t * 16);
        store_row0_h16_relu(c, xbuf, t * 16, head_b1);
    }
    __syncthreads();
    // @ head_w2, relu -> zA (f32)
    for (int t = wave; t < 16; t += 8) {
        v8f c = gemm_tile_p(xbuf, XS, wsp + OFF_HW2, 256, t * 16);
        store_row0_f32_relu(c, zA, t * 16, head_b2);
    }
    __syncthreads();
    // final dot with head_w3 [256] + b3
    {
        float p = zA[tid] * head_w3[tid];
        #pragma unroll
        for (int m = 1; m < 32; m <<= 1) p += __shfl_xor(p, m, 32);
        if ((tid & 31) == 0) red[wave] = p;
        __syncthreads();
        if (tid == 0) {
            float t = 0.0f;
            for (int w = 0; w < 8; ++w) t += red[w];
            outp[n] = t + head_b3[0];
        }
    }
}

extern "C" void kernel_launch(void* const* d_in, const int* in_sizes, int n_in,
                              void* d_out, int out_size, void* d_ws, size_t ws_size,
                              hipStream_t stream) {
    (void)n_in; (void)out_size; (void)ws_size;
    const float* img_feat  = (const float*)d_in[0];
    const float* pts_view  = (const float*)d_in[1];
    const float* pts_world = (const float*)d_in[2];
    const float* view_w    = (const float*)d_in[3];
    const float* view_b    = (const float*)d_in[4];
    const float* cls_token = (const float*)d_in[5];
    const float* cls_w     = (const float*)d_in[6];
    const float* cls_b     = (const float*)d_in[7];
    const float* pos_emb   = (const float*)d_in[8];
    const float* ln1_g     = (const float*)d_in[9];
    const float* ln1_b     = (const float*)d_in[10];
    const float* qkv_w     = (const float*)d_in[11];
    const float* reattn_w  = (const float*)d_in[12];
    const float* reattn_g  = (const float*)d_in[13];
    const float* reattn_b  = (const float*)d_in[14];
    const float* out_w     = (const float*)d_in[15];
    const float* out_b     = (const float*)d_in[16];
    const float* ln2_g     = (const float*)d_in[17];
    const float* ln2_b     = (const float*)d_in[18];
    const float* ff_w1     = (const float*)d_in[19];
    const float* ff_b1     = (const float*)d_in[20];
    const float* ff_w2     = (const float*)d_in[21];
    const float* ff_b2     = (const float*)d_in[22];
    const float* head_ln_g = (const float*)d_in[23];
    const float* head_ln_b = (const float*)d_in[24];
    const float* head_w1   = (const float*)d_in[25];
    const float* head_b1   = (const float*)d_in[26];
    const float* head_w2   = (const float*)d_in[27];
    const float* head_b2   = (const float*)d_in[28];
    const float* head_w3   = (const float*)d_in[29];
    const float* head_b3   = (const float*)d_in[30];

    _Float16* wsp = (_Float16*)d_ws;
    const int N = in_sizes[0] / (8 * 256);   // img_feat is [N,8,256]

    auto prep = [&](const float* src, _Float16* dst, int K, int Nc, int nmat,
                    size_t sstr, size_t dstr) {
        int ntiles = (K >> 5) * (Nc >> 4);
        dim3 g((ntiles * 32 + 255) / 256, nmat);
        prep_weights<<<g, 256, 0, stream>>>(src, dst, K, Nc, sstr, dstr);
    };
    prep(view_w,  wsp + OFF_VIEW, 320, 256,  1, 0, 0);
    prep(cls_w,   wsp + OFF_CLS,  320, 256,  1, 0, 0);
    prep(qkv_w,   wsp + OFF_QKV,  256, 1536, 4, (size_t)256 * 1536, 393216);
    prep(out_w,   wsp + OFF_OUT,  512, 256,  4, (size_t)512 * 256,  131072);
    prep(ff_w1,   wsp + OFF_FF1,  256, 1024, 4, (size_t)256 * 1024, 262144);
    prep(ff_w2,   wsp + OFF_FF2,  1024, 256, 4, (size_t)1024 * 256, 262144);
    prep(head_w1, wsp + OFF_HW1,  256, 256,  1, 0, 0);
    prep(head_w2, wsp + OFF_HW2,  256, 256,  1, 0, 0);

    occvit_fused<<<N, 256, 0, stream>>>(
        img_feat, pts_view, pts_world, view_b, cls_token, cls_b, pos_emb,
        ln1_g, ln1_b, reattn_w, reattn_g, reattn_b, out_b, ln2_g, ln2_b,
        ff_b1, ff_b2, head_ln_g, head_ln_b, head_b1, head_b2,
        head_w3, head_b3, wsp, (float*)d_out);
}
